// GAT_57312043598042
// MI455X (gfx1250) — matrix operand order repaired and checked by reading, not compiled, not run on hardware
//
#include <hip/hip_runtime.h>

typedef __attribute__((ext_vector_type(16))) _Float16 v16h;
typedef __attribute__((ext_vector_type(8)))  float    v8f;

#define D_IN  128
#define D_HID 64
#define D_OUTC 6
#define NEG_SLOPE 0.2f

// ---------- helpers ----------
__device__ __forceinline__ unsigned ford(float f) {
  unsigned u = __float_as_uint(f);
  return (u & 0x80000000u) ? ~u : (u | 0x80000000u);
}
__device__ __forceinline__ float fuord(unsigned u) {
  return (u & 0x80000000u) ? __uint_as_float(u ^ 0x80000000u) : __uint_as_float(~u);
}
__device__ __forceinline__ float lrelu(float x) { return x > 0.f ? x : NEG_SLOPE * x; }

__device__ __forceinline__ void edge_ids(int k, const int* __restrict__ src,
                                         const int* __restrict__ dst, int E,
                                         int& a, int& b) {
  if (k < E) { a = src[k]; b = dst[k]; } else { a = b = k - E; }
}

// Convert 4 raw float4s (two contiguous 8-float runs) into the ISA 16x32 f16
// A-fragment layout for one k-tile:
//   a[0..7]  <- K = k0 + half*8 + i         (r0, r1)
//   a[8..15] <- K = k0 + 16 + half*8 + i    (r2, r3)
__device__ __forceinline__ v16h cvt_afrag(float4 r0, float4 r1, float4 r2, float4 r3) {
  v16h a;
  a[0]  = (_Float16)r0.x; a[1]  = (_Float16)r0.y; a[2]  = (_Float16)r0.z; a[3]  = (_Float16)r0.w;
  a[4]  = (_Float16)r1.x; a[5]  = (_Float16)r1.y; a[6]  = (_Float16)r1.z; a[7]  = (_Float16)r1.w;
  a[8]  = (_Float16)r2.x; a[9]  = (_Float16)r2.y; a[10] = (_Float16)r2.z; a[11] = (_Float16)r2.w;
  a[12] = (_Float16)r3.x; a[13] = (_Float16)r3.y; a[14] = (_Float16)r3.z; a[15] = (_Float16)r3.w;
  return a;
}

// ---------- GEMM1: h1[N,64] = f16(x[N,128]) @ f16(W1[128,64]), f32 accum ----------
#define WT1_STRIDE 136   // 64 rows x 136 f16 (padded) -> conflict-free lane reads
__global__ __launch_bounds__(256) void gemm1_kernel(const float* __restrict__ x,
                                                    const float* __restrict__ W,
                                                    float* __restrict__ h, int N) {
  __shared__ _Float16 wt[64 * WT1_STRIDE];   // wt[n][k] = W[k][n]
  int t = threadIdx.x;
  for (int i = t; i < 64 * 128; i += 256) {
    int n = i >> 7, k = i & 127;
    wt[n * WT1_STRIDE + k] = (_Float16)W[k * 64 + n];
  }

  int wave = t >> 5, lane = t & 31;
  int half = lane >> 4;                // 0: lanes 0-15, 1: lanes 16-31
  int rowBase = (blockIdx.x * 8 + wave) * 16;
  int mrow = rowBase + (lane & 15);
  bool rowOK = mrow < N;
  const float* xr = x + (size_t)(rowOK ? mrow : 0) * D_IN;  // clamped: loads always safe

  // Issue ALL A-fragment global loads up front (16 x b128 in flight).
  float4 raw[16];
#pragma unroll
  for (int kt = 0; kt < 4; ++kt) {
    const float4* p = (const float4*)(xr + kt * 32 + (half << 3));
    raw[kt * 4 + 0] = p[0];
    raw[kt * 4 + 1] = p[1];
    raw[kt * 4 + 2] = p[4];   // +16 floats
    raw[kt * 4 + 3] = p[5];
  }
  v16h afrag[4];
#pragma unroll
  for (int kt = 0; kt < 4; ++kt)
    afrag[kt] = cvt_afrag(raw[kt * 4 + 0], raw[kt * 4 + 1], raw[kt * 4 + 2], raw[kt * 4 + 3]);

  __syncthreads();

  v8f acc0 = {}, acc1 = {}, acc2 = {}, acc3 = {};
  int n = lane & 15;
#pragma unroll
  for (int kt = 0; kt < 4; ++kt) {
    int kb = kt * 32 + (half << 3);                            // ISA B-frag layout
    v16h b0, b1, b2, b3;
#pragma unroll
    for (int i = 0; i < 16; ++i) {
      b0[i] = wt[(n +  0) * WT1_STRIDE + kb + (i & 7) + ((i >> 3) << 4)];
      b1[i] = wt[(n + 16) * WT1_STRIDE + kb + (i & 7) + ((i >> 3) << 4)];
      b2[i] = wt[(n + 32) * WT1_STRIDE + kb + (i & 7) + ((i >> 3) << 4)];
      b3[i] = wt[(n + 48) * WT1_STRIDE + kb + (i & 7) + ((i >> 3) << 4)];
    }
    acc0 = __builtin_amdgcn_wmma_f32_16x16x32_f16(false, afrag[kt], false, b0, (short)0, acc0, false, false);
    acc1 = __builtin_amdgcn_wmma_f32_16x16x32_f16(false, afrag[kt], false, b1, (short)0, acc1, false, false);
    acc2 = __builtin_amdgcn_wmma_f32_16x16x32_f16(false, afrag[kt], false, b2, (short)0, acc2, false, false);
    acc3 = __builtin_amdgcn_wmma_f32_16x16x32_f16(false, afrag[kt], false, b3, (short)0, acc3, false, false);
  }
  int col = lane & 15;
#pragma unroll
  for (int r = 0; r < 8; ++r) {
    int row = rowBase + r + (half << 3);
    if (row < N) {
      float* o = h + (size_t)row * D_HID + col;
      o[0]  = acc0[r];
      o[16] = acc1[r];
      o[32] = acc2[r];
      o[48] = acc3[r];
    }
  }
}

// ---------- GEMM2: h2[N,6] = f16(hact[N,64]) @ f16(W2[64,6]) ----------
#define WT2_STRIDE 72
__global__ __launch_bounds__(256) void gemm2_kernel(const float* __restrict__ hact,
                                                    const float* __restrict__ W,
                                                    float* __restrict__ h2, int N) {
  __shared__ _Float16 wt[16 * WT2_STRIDE];   // padded to 16 cols, zeros for n>=6
  int t = threadIdx.x;
  for (int i = t; i < 16 * 64; i += 256) {
    int n = i >> 6, k = i & 63;
    wt[n * WT2_STRIDE + k] = (n < D_OUTC) ? (_Float16)W[k * D_OUTC + n] : (_Float16)0.f;
  }

  int wave = t >> 5, lane = t & 31;
  int half = lane >> 4;
  int rowBase = (blockIdx.x * 8 + wave) * 16;
  int mrow = rowBase + (lane & 15);
  bool rowOK = mrow < N;
  const float* xr = hact + (size_t)(rowOK ? mrow : 0) * D_HID;  // clamped

  float4 raw[8];
#pragma unroll
  for (int kt = 0; kt < 2; ++kt) {
    const float4* p = (const float4*)(xr + kt * 32 + (half << 3));
    raw[kt * 4 + 0] = p[0];
    raw[kt * 4 + 1] = p[1];
    raw[kt * 4 + 2] = p[4];
    raw[kt * 4 + 3] = p[5];
  }
  v16h afrag[2];
#pragma unroll
  for (int kt = 0; kt < 2; ++kt)
    afrag[kt] = cvt_afrag(raw[kt * 4 + 0], raw[kt * 4 + 1], raw[kt * 4 + 2], raw[kt * 4 + 3]);

  __syncthreads();

  v8f acc = {};
  int n = lane & 15;
#pragma unroll
  for (int kt = 0; kt < 2; ++kt) {
    int kb = kt * 32 + (half << 3);
    v16h b;
#pragma unroll
    for (int i = 0; i < 16; ++i) b[i] = wt[n * WT2_STRIDE + kb + (i & 7) + ((i >> 3) << 4)];
    acc = __builtin_amdgcn_wmma_f32_16x16x32_f16(false, afrag[kt], false, b, (short)0, acc, false, false);
  }
  int col = lane & 15;
  if (col < D_OUTC) {
#pragma unroll
    for (int r = 0; r < 8; ++r) {
      int row = rowBase + r + (half << 3);
      if (row < N) h2[(size_t)row * D_OUTC + col] = acc[r];
    }
  }
}

// ---------- per-node attention dot products ----------
__global__ void dots64_kernel(const float* __restrict__ h, const float* __restrict__ as,
                              const float* __restrict__ ad, float* __restrict__ s,
                              float* __restrict__ tt, int N) {
  int n = blockIdx.x * blockDim.x + threadIdx.x;
  if (n >= N) return;
  const float4* r4 = (const float4*)(h + (size_t)n * D_HID);
  const float4* a4 = (const float4*)as;
  const float4* d4 = (const float4*)ad;
  float a = 0.f, b = 0.f;
#pragma unroll
  for (int c = 0; c < D_HID / 4; ++c) {
    float4 v = r4[c], av = a4[c], dv = d4[c];
    a += v.x * av.x + v.y * av.y + v.z * av.z + v.w * av.w;
    b += v.x * dv.x + v.y * dv.y + v.z * dv.z + v.w * dv.w;
  }
  s[n] = a; tt[n] = b;
}

__global__ void dots6_kernel(const float* __restrict__ h, const float* __restrict__ as,
                             const float* __restrict__ ad, float* __restrict__ s,
                             float* __restrict__ tt, int N) {
  int n = blockIdx.x * blockDim.x + threadIdx.x;
  if (n >= N) return;
  const float* r = h + (size_t)n * D_OUTC;
  float a = 0.f, b = 0.f;
#pragma unroll
  for (int c = 0; c < D_OUTC; ++c) { float v = r[c]; a += v * as[c]; b += v * ad[c]; }
  s[n] = a; tt[n] = b;
}

// ---------- init kernels ----------
__global__ void zero_f32(float* p, long long n) {
  long long i = blockIdx.x * (long long)blockDim.x + threadIdx.x;
  if (i < n) p[i] = 0.f;
}
__global__ void zero_u32(unsigned* p, int n) {
  int i = blockIdx.x * blockDim.x + threadIdx.x;
  if (i < n) p[i] = 0u;
}
__global__ void init_out_kernel(float* __restrict__ out, const float* __restrict__ b2, int N) {
  int i = blockIdx.x * blockDim.x + threadIdx.x;
  if (i < N * D_OUTC) out[i] = b2[i % D_OUTC];
}

// ---------- edge softmax passes ----------
__global__ void edge_max_kernel(const int* __restrict__ src, const int* __restrict__ dst,
                                const float* __restrict__ s, const float* __restrict__ t,
                                unsigned* __restrict__ m, int E, int N) {
  int k = blockIdx.x * blockDim.x + threadIdx.x;
  if (k >= E + N) return;
  int a, b; edge_ids(k, src, dst, E, a, b);
  float e = lrelu(s[a] + t[b]);
  atomicMax(&m[b], ford(e));
}

__global__ void edge_expsum_kernel(const int* __restrict__ src, const int* __restrict__ dst,
                                   const float* __restrict__ s, const float* __restrict__ t,
                                   const unsigned* __restrict__ m, float* __restrict__ ex,
                                   float* __restrict__ den, int E, int N) {
  int k = blockIdx.x * blockDim.x + threadIdx.x;
  if (k >= E + N) return;
  int a, b; edge_ids(k, src, dst, E, a, b);
  float e = lrelu(s[a] + t[b]);
  float xv = __expf(e - fuord(m[b]));
  ex[k] = xv;
  atomicAdd(&den[b], xv);
}

// ---------- edge aggregation: 16 threads/edge, 4 channels each (layer 1) ----------
__global__ void edge_agg64_kernel(const int* __restrict__ src, const int* __restrict__ dst,
                                  const float* __restrict__ ex, const float* __restrict__ den,
                                  const float* __restrict__ h, float* __restrict__ agg,
                                  int E, int N) {
  long long gid = blockIdx.x * (long long)blockDim.x + threadIdx.x;
  int k = (int)(gid >> 4);
  int sub = (int)(gid & 15);
  if (k >= E + N) return;
  int a, b; edge_ids(k, src, dst, E, a, b);
  float alpha = ex[k] / den[b];
  const float4 hv = *(const float4*)(h + (size_t)a * D_HID + sub * 4);
  float* o = agg + (size_t)b * D_HID + sub * 4;
  atomicAdd(o + 0, hv.x * alpha);
  atomicAdd(o + 1, hv.y * alpha);
  atomicAdd(o + 2, hv.z * alpha);
  atomicAdd(o + 3, hv.w * alpha);
}

// ---------- edge aggregation: 8 threads/edge, 1 channel each (layer 2) ----------
__global__ void edge_agg6_kernel(const int* __restrict__ src, const int* __restrict__ dst,
                                 const float* __restrict__ ex, const float* __restrict__ den,
                                 const float* __restrict__ h, float* __restrict__ out,
                                 int E, int N) {
  long long gid = blockIdx.x * (long long)blockDim.x + threadIdx.x;
  int k = (int)(gid >> 3);
  int sub = (int)(gid & 7);
  if (k >= E + N || sub >= D_OUTC) return;
  int a, b; edge_ids(k, src, dst, E, a, b);
  float alpha = ex[k] / den[b];
  atomicAdd(out + (size_t)b * D_OUTC + sub, h[(size_t)a * D_OUTC + sub] * alpha);
}

// ---------- bias + ELU ----------
__global__ void bias_elu_kernel(const float* __restrict__ agg, const float* __restrict__ bias,
                                float* __restrict__ o, long long total) {
  long long i = blockIdx.x * (long long)blockDim.x + threadIdx.x;
  if (i >= total) return;
  float v = agg[i] + bias[i & (D_HID - 1)];
  o[i] = v > 0.f ? v : expm1f(v);
}

// ---------- host launcher ----------
extern "C" void kernel_launch(void* const* d_in, const int* in_sizes, int n_in,
                              void* d_out, int out_size, void* d_ws, size_t ws_size,
                              hipStream_t stream) {
  (void)n_in; (void)out_size; (void)ws_size;
  const float* x   = (const float*)d_in[0];
  const int*  eidx = (const int*)d_in[1];
  const float* W1  = (const float*)d_in[2];
  const float* as1 = (const float*)d_in[3];
  const float* ad1 = (const float*)d_in[4];
  const float* b1  = (const float*)d_in[5];
  const float* W2  = (const float*)d_in[6];
  const float* as2 = (const float*)d_in[7];
  const float* ad2 = (const float*)d_in[8];
  const float* b2  = (const float*)d_in[9];
  float* out = (float*)d_out;

  int N = in_sizes[0] / D_IN;
  int E = in_sizes[1] / 2;
  const int* src = eidx;
  const int* dst = eidx + E;
  int EN = E + N;

  // workspace carve-up (floats)
  float* ws = (float*)d_ws;
  size_t N64 = (size_t)N * D_HID;
  float* h1   = ws; ws += N64;
  float* agg1 = ws; ws += N64;
  float* s1   = ws; ws += N;
  float* t1   = ws; ws += N;
  float* s2   = ws; ws += N;
  float* t2   = ws; ws += N;
  unsigned* mb = (unsigned*)ws; ws += N;
  float* den  = ws; ws += N;
  float* ex   = ws; ws += (size_t)EN;
  float* h2   = ws; ws += (size_t)N * D_OUTC;

  dim3 blk(256);
  int gRows = (N + 127) / 128;
  int gN    = (N + 255) / 256;
  int gEN   = (EN + 255) / 256;
  long long n64ll = (long long)N64;
  int gN64  = (int)((n64ll + 255) / 256);

  // ---- layer 1 ----
  gemm1_kernel<<<gRows, blk, 0, stream>>>(x, W1, h1, N);
  dots64_kernel<<<gN, blk, 0, stream>>>(h1, as1, ad1, s1, t1, N);
  zero_u32<<<gN, blk, 0, stream>>>(mb, N);
  zero_f32<<<gN, blk, 0, stream>>>(den, (long long)N);
  edge_max_kernel<<<gEN, blk, 0, stream>>>(src, dst, s1, t1, mb, E, N);
  edge_expsum_kernel<<<gEN, blk, 0, stream>>>(src, dst, s1, t1, mb, ex, den, E, N);
  zero_f32<<<gN64, blk, 0, stream>>>(agg1, n64ll);
  long long a1threads = (long long)EN * 16;
  edge_agg64_kernel<<<(int)((a1threads + 255) / 256), blk, 0, stream>>>(src, dst, ex, den, h1, agg1, E, N);
  bias_elu_kernel<<<gN64, blk, 0, stream>>>(agg1, b1, h1, n64ll);  // h1 becomes elu(agg1+b1)

  // ---- layer 2 ----
  gemm2_kernel<<<gRows, blk, 0, stream>>>(h1, W2, h2, N);
  dots6_kernel<<<gN, blk, 0, stream>>>(h2, as2, ad2, s2, t2, N);
  zero_u32<<<gN, blk, 0, stream>>>(mb, N);
  zero_f32<<<gN, blk, 0, stream>>>(den, (long long)N);
  edge_max_kernel<<<gEN, blk, 0, stream>>>(src, dst, s2, t2, mb, E, N);
  edge_expsum_kernel<<<gEN, blk, 0, stream>>>(src, dst, s2, t2, mb, ex, den, E, N);
  init_out_kernel<<<(N * D_OUTC + 255) / 256, blk, 0, stream>>>(out, b2, N);
  long long a2threads = (long long)EN * 8;
  edge_agg6_kernel<<<(int)((a2threads + 255) / 256), blk, 0, stream>>>(src, dst, ex, den, h2, out, E, N);
}
